// AttentionPairBias_4724464026005
// MI455X (gfx1250) — matrix operand order repaired and checked by reading, not compiled, not run on hardware
//
#include <hip/hip_runtime.h>

// ---------------------------------------------------------------------------
// Problem constants (B=1)
// ---------------------------------------------------------------------------
#define N_TOK 4096
#define CA    768
#define CS    384
#define CZ    128
#define NH    16
#define NQ    32
#define NKY   128
#define NB    128   // N_TOK / NQ
#define DH    48    // CA / NH
#define DP    64    // padded head dim for WMMA (K multiple of 32)
#define KVP   72    // LDS row pitch for K/V tiles (64 + 8 pad -> conflict-free)

#ifndef __has_builtin
#define __has_builtin(x) 0
#endif
#if __has_builtin(__builtin_amdgcn_tensor_load_to_lds) && __has_builtin(__builtin_amdgcn_s_wait_tensorcnt)
#define TDM_AVAILABLE 1
#else
#define TDM_AVAILABLE 0
#endif

typedef __attribute__((ext_vector_type(16))) __bf16 bfrag;
typedef __attribute__((ext_vector_type(8)))  float  v8f;
typedef __attribute__((ext_vector_type(4)))  unsigned uint32x4;
typedef __attribute__((ext_vector_type(8)))  int      int32x8;
typedef __attribute__((ext_vector_type(4)))  int      int32x4;

union FragU { unsigned u[8]; bfrag b; };

__device__ __forceinline__ unsigned short f2bf(float f) {
    unsigned u = __builtin_bit_cast(unsigned, f);
    u += 0x7FFFu + ((u >> 16) & 1u);          // round-to-nearest-even
    return (unsigned short)(u >> 16);
}

__device__ __forceinline__ float sigm(float x) { return 1.f / (1.f + __expf(-x)); }

__device__ __forceinline__ v8f wmma_bf16(bfrag a, bfrag b, v8f c) {
    return __builtin_amdgcn_wmma_f32_16x16x32_bf16(
        false, a, false, b, (short)0, c, false, false);
}

// A-matrix fragment (16x32 bf16, M = lane%16), contiguous k = 16*half + element.
__device__ __forceinline__ bfrag load_frag_A(const unsigned short* rowptr, int kk, int lane) {
    FragU f; int half = lane >> 4;
#pragma unroll
    for (int v = 0; v < 8; ++v) {
        int k = kk + half * 16 + v * 2;
        f.u[v] = *(const unsigned*)(rowptr + k);
    }
    return f.b;
}

// B-matrix fragment (32x16 bf16, N = lane%16) from an N-major buffer.
__device__ __forceinline__ bfrag load_frag_B(const unsigned short* colptr, int kk, int lane) {
    FragU f; int half = lane >> 4;
#pragma unroll
    for (int v = 0; v < 8; ++v) {
        int k = kk + half * 16 + 2 * v;
        f.u[v] = *(const unsigned*)(colptr + k);
    }
    return f.b;
}

#if TDM_AVAILABLE
// Issue a 2D TDM load: tile_d1 rows of tile_d0 bf16 elements, global row stride
// row_stride_elems, into LDS at lds_off, with optional LDS row padding.
// D# layout per CDNA5 ISA 8.3-8.6.
__device__ __forceinline__ void tdm_load_2d(unsigned lds_off, const void* gaddr,
                                            unsigned tile_d0, unsigned tile_d1,
                                            unsigned row_stride_elems,
                                            unsigned pad_interval_code,
                                            unsigned pad_amount_code, bool pad_en) {
    unsigned long long ga = (unsigned long long)gaddr;
    uint32x4 g0;
    g0[0] = 1u;                                          // count=1, user load
    g0[1] = lds_off;                                     // lds_addr
    g0[2] = (unsigned)(ga & 0xFFFFFFFFu);                // global_addr[31:0]
    g0[3] = (unsigned)((ga >> 32) & 0x1FFFFFFu) | (2u << 30);  // addr[56:32] | type=2
    unsigned w[8];
    w[0] = (1u << 16) | (pad_en ? (1u << 20) : 0u)
         | (pad_interval_code << 22) | (pad_amount_code << 25); // data_size=2B
    w[1] = (tile_d0 & 0xFFFFu) << 16;                    // tensor_dim0[15:0] @48
    w[2] = ((tile_d0 >> 16) & 0xFFFFu) | ((tile_d1 & 0xFFFFu) << 16); // dim0 hi, dim1 lo
    w[3] = ((tile_d1 >> 16) & 0xFFFFu) | ((tile_d0 & 0xFFFFu) << 16); // dim1 hi, tile_dim0
    w[4] = (tile_d1 & 0xFFFFu);                          // tile_dim1 (tile_dim2=0)
    w[5] = row_stride_elems;                             // tensor_dim0_stride lo32
    w[6] = 0u;                                           // stride0 hi16, stride1 lo16
    w[7] = 0u;
    int32x8 g1;
#pragma unroll
    for (int i = 0; i < 8; ++i) g1[i] = (int)w[i];
    int32x4 gz4 = {0, 0, 0, 0};
#if __clang_major__ >= 23
    int32x8 gz8 = {0, 0, 0, 0, 0, 0, 0, 0};
    __builtin_amdgcn_tensor_load_to_lds(g0, g1, gz4, gz4, gz8, 0);
#else
    __builtin_amdgcn_tensor_load_to_lds(g0, g1, gz4, gz4, 0);
#endif
}
#endif

// ---------------------------------------------------------------------------
// Prep: transpose f32 [K][N] -> bf16 [N][K]
// ---------------------------------------------------------------------------
__global__ void k_transpose_bf16(const float* __restrict__ src, unsigned short* __restrict__ dst,
                                 int K, int Ncols) {
    int i = blockIdx.x * blockDim.x + threadIdx.x;
    if (i < K * Ncols) {
        int k = i / Ncols, n = i % Ncols;
        dst[(long)n * K + k] = f2bf(src[i]);
    }
}

// ---------------------------------------------------------------------------
// LayerNorm of s -> s_ln (bf16) ; also raw s -> bf16
// ---------------------------------------------------------------------------
__global__ void k_ln_s(const float* __restrict__ s, unsigned short* __restrict__ sln_bf,
                       unsigned short* __restrict__ s_bf) {
    int row = blockIdx.x, t = threadIdx.x;                // 128 threads
    const float* sr = s + (long)row * CS;
    float x[3]; float sum = 0.f;
#pragma unroll
    for (int j = 0; j < 3; ++j) { x[j] = sr[t + j * 128]; sum += x[j]; }
    __shared__ float red[128];
    __shared__ float mu_s, rs_s;
    red[t] = sum; __syncthreads();
    for (int off = 64; off > 0; off >>= 1) { if (t < off) red[t] += red[t + off]; __syncthreads(); }
    if (t == 0) mu_s = red[0] / (float)CS;
    __syncthreads();
    float mu = mu_s, vs = 0.f;
#pragma unroll
    for (int j = 0; j < 3; ++j) { float d = x[j] - mu; vs += d * d; }
    red[t] = vs; __syncthreads();
    for (int off = 64; off > 0; off >>= 1) { if (t < off) red[t] += red[t + off]; __syncthreads(); }
    if (t == 0) rs_s = rsqrtf(red[0] / (float)CS + 1e-5f);
    __syncthreads();
    float rs = rs_s;
#pragma unroll
    for (int j = 0; j < 3; ++j) {
        int c = t + j * 128;
        sln_bf[(long)row * CS + c] = f2bf((x[j] - mu) * rs);
        s_bf[(long)row * CS + c]   = f2bf(x[j]);
    }
}

// LayerNorm of a -> a_ln (f32)
__global__ void k_ln_a(const float* __restrict__ a, float* __restrict__ aln) {
    int row = blockIdx.x, t = threadIdx.x;                // 256 threads
    const float* ar = a + (long)row * CA;
    float x[3]; float sum = 0.f;
#pragma unroll
    for (int j = 0; j < 3; ++j) { x[j] = ar[t + j * 256]; sum += x[j]; }
    __shared__ float red[256];
    __shared__ float mu_s, rs_s;
    red[t] = sum; __syncthreads();
    for (int off = 128; off > 0; off >>= 1) { if (t < off) red[t] += red[t + off]; __syncthreads(); }
    if (t == 0) mu_s = red[0] / (float)CA;
    __syncthreads();
    float mu = mu_s, vs = 0.f;
#pragma unroll
    for (int j = 0; j < 3; ++j) { float d = x[j] - mu; vs += d * d; }
    red[t] = vs; __syncthreads();
    for (int off = 128; off > 0; off >>= 1) { if (t < off) red[t] += red[t + off]; __syncthreads(); }
    if (t == 0) rs_s = rsqrtf(red[0] / (float)CA + 1e-5f);
    __syncthreads();
    float rs = rs_s;
#pragma unroll
    for (int j = 0; j < 3; ++j) aln[(long)row * CA + t + j * 256] = (x[j] - mu) * rs;
}

// ---------------------------------------------------------------------------
// adaLN: a2 = sigmoid(s_ln@Wg + gb) * a_ln + s_ln@Ws
// Each wave: one 16-row M-tile x 3 N-tiles, A fragment reused for 6 WMMAs/k-step.
// ---------------------------------------------------------------------------
__global__ void k_adaln(const unsigned short* __restrict__ sln,      // [N][CS] bf16
                        const unsigned short* __restrict__ wgT,      // [CA][CS] bf16
                        const unsigned short* __restrict__ wsT,      // [CA][CS] bf16
                        const float* __restrict__ gb,
                        const float* __restrict__ aln,               // [N][CA] f32
                        unsigned short* __restrict__ a2_bf) {        // [N][CA] bf16
    int lane = threadIdx.x & 31, wave = threadIdx.x >> 5;
    int job = blockIdx.x * 8 + wave;          // 4096 jobs = 256 Mtiles x 16 groups
    int tm = job >> 4, tg = job & 15;         // 16 groups of 3 N-tiles
    int half = lane >> 4, ln = lane & 15;
    long row = (long)tm * 16 + ln;
    const unsigned short* arow = sln + row * CS;
    v8f accg[3] = {}, accs[3] = {};
    for (int kk = 0; kk < CS; kk += 32) {
        bfrag af = load_frag_A(arow, kk, lane);
#pragma unroll
        for (int t = 0; t < 3; ++t) {
            long col = (long)(tg * 3 + t) * 16 + ln;
            accg[t] = wmma_bf16(af, load_frag_B(wgT + col * CS, kk, lane), accg[t]);
            accs[t] = wmma_bf16(af, load_frag_B(wsT + col * CS, kk, lane), accs[t]);
        }
    }
#pragma unroll
    for (int t = 0; t < 3; ++t) {
        int col = (tg * 3 + t) * 16 + ln;
        float gbias = gb[col];
#pragma unroll
        for (int j = 0; j < 8; ++j) {
            long r = (long)tm * 16 + j + half * 8;
            float gate = sigm(accg[t][j] + gbias);
            a2_bf[r * CA + col] = f2bf(gate * aln[r * CA + col] + accs[t][j]);
        }
    }
}

// ---------------------------------------------------------------------------
// q/k/v/g projections. mode = blockIdx.y: 0=q (scaled), 1=k, 2=v, 3=g
// Each wave: 16 rows x 6 N-tiles (A fragment reused 6x).
// ---------------------------------------------------------------------------
__global__ void k_qkvg(const unsigned short* __restrict__ a2,        // [N][CA] bf16
                       const unsigned short* __restrict__ wqT,
                       const unsigned short* __restrict__ wkT,
                       const unsigned short* __restrict__ wvT,
                       const unsigned short* __restrict__ wgT,
                       const float* __restrict__ bgb,
                       unsigned short* __restrict__ qp,
                       unsigned short* __restrict__ kp,
                       unsigned short* __restrict__ vp,
                       float* __restrict__ gout) {
    int lane = threadIdx.x & 31, wave = threadIdx.x >> 5;
    int job = blockIdx.x * 8 + wave;          // 2048 jobs = 256 Mtiles x 8 groups
    int tm = job >> 3, tg = job & 7;          // 8 groups of 6 N-tiles
    int mode = blockIdx.y;
    int half = lane >> 4, ln = lane & 15;
    long row = (long)tm * 16 + ln;
    const unsigned short* wT = (mode == 0) ? wqT : (mode == 1) ? wkT : (mode == 2) ? wvT : wgT;
    const unsigned short* arow = a2 + row * CA;
    v8f acc[6] = {};
    for (int kk = 0; kk < CA; kk += 32) {
        bfrag af = load_frag_A(arow, kk, lane);
#pragma unroll
        for (int t = 0; t < 6; ++t)
            acc[t] = wmma_bf16(af, load_frag_B(wT + (long)((tg * 6 + t) * 16 + ln) * CA, kk, lane),
                               acc[t]);
    }
    const float qscale = 0.14433756729740643f;   // 1/sqrt(48)
#pragma unroll
    for (int t = 0; t < 6; ++t) {
        int col = (tg * 6 + t) * 16 + ln;
        int h = col / DH, d = col % DH;          // 16-wide tile stays inside one head
#pragma unroll
        for (int j = 0; j < 8; ++j) {
            long r = (long)tm * 16 + j + half * 8;
            float v = acc[t][j];
            if (mode == 0)      qp[(r * NH + h) * DP + d] = f2bf(v * qscale);
            else if (mode == 1) kp[(r * NH + h) * DP + d] = f2bf(v);
            else if (mode == 2) vp[(r * NH + h) * DP + d] = f2bf(v);
            else                gout[r * CA + col] = sigm(v + bgb[col]);
        }
    }
}

// ---------------------------------------------------------------------------
// Pair bias: LN(z;gz,bz) @ wz  -> bias[nb][h][q][k]   (WMMA projection)
// ---------------------------------------------------------------------------
__global__ void k_bias(const float* __restrict__ z, const float* __restrict__ gz,
                       const float* __restrict__ bz, const unsigned short* __restrict__ wzT,
                       float* __restrict__ bias) {
    __shared__ float red[16][17];
    __shared__ float musig[16][2];
    __shared__ unsigned short zl[16][CZ];
    int t = threadIdx.x;                       // 256 threads
    int r = t >> 4, lb = t & 15;
    long R0 = (long)blockIdx.x * 16;
    const float* zr = z + (R0 + r) * CZ + lb * 8;
    float x[8]; float sum = 0.f;
#pragma unroll
    for (int j = 0; j < 8; ++j) { x[j] = zr[j]; sum += x[j]; }
    red[r][lb] = sum; __syncthreads();
    if (lb == 0) { float s2 = 0.f; for (int i = 0; i < 16; ++i) s2 += red[r][i];
                   musig[r][0] = s2 / (float)CZ; }
    __syncthreads();
    float mu = musig[r][0], vs = 0.f;
#pragma unroll
    for (int j = 0; j < 8; ++j) { float d = x[j] - mu; vs += d * d; }
    red[r][lb] = vs; __syncthreads();
    if (lb == 0) { float s2 = 0.f; for (int i = 0; i < 16; ++i) s2 += red[r][i];
                   musig[r][1] = rsqrtf(s2 / (float)CZ + 1e-5f); }
    __syncthreads();
    float rs = musig[r][1];
#pragma unroll
    for (int j = 0; j < 8; ++j) {
        int c = lb * 8 + j;
        zl[r][c] = f2bf((x[j] - mu) * rs * gz[c] + bz[c]);
    }
    __syncthreads();
    if (t < 32) {                              // wave 0: 16x16x128 projection
        int lane = t, half = lane >> 4, ln = lane & 15;
        v8f acc = {};
        const unsigned short* arow = &zl[ln][0];
        const unsigned short* bcol = wzT + (long)ln * CZ;
#pragma unroll
        for (int kk = 0; kk < CZ; kk += 32)
            acc = wmma_bf16(load_frag_A(arow, kk, lane), load_frag_B(bcol, kk, lane), acc);
        int nb = (int)(R0 >> 12), rem = (int)(R0 & 4095);
        int q = rem >> 7, k0 = rem & 127;
#pragma unroll
        for (int j = 0; j < 8; ++j) {
            int M = j + half * 8;
            bias[(((long)nb * NH + ln) * NQ + q) * NKY + (k0 + M)] = acc[j];
        }
    }
}

// ---------------------------------------------------------------------------
// Windowed attention. grid = (NB, NH), block = 64 (2 waves; 16 queries each).
// K/V window staged into LDS once per block (TDM if available).
// ---------------------------------------------------------------------------
__global__ void k_attn(const unsigned short* __restrict__ qp,
                       const unsigned short* __restrict__ kp,
                       const unsigned short* __restrict__ vp,
                       const float* __restrict__ bias,
                       float* __restrict__ o) {
    int nb = blockIdx.x, h = blockIdx.y;
    int lane = threadIdx.x & 31, mt = threadIdx.x >> 5;
    int half = lane >> 4, ln = lane & 15;
    __shared__ unsigned short Kt[NKY][KVP];
    __shared__ unsigned short Vt[NKY][KVP];
    __shared__ unsigned short P[2][16][NKY];

    // valid key range: token = nb*NQ + key - 48 in [0, N_TOK)
    int lo = 48 - nb * NQ;            if (lo < 0) lo = 0;
    int hi = N_TOK + 48 - nb * NQ;    if (hi > NKY) hi = NKY;

    // zero invalid LDS rows (avoid NaN in P@V)
    for (int rr = 0; rr < NKY; ++rr) {
        if (rr < lo || rr >= hi) {
            for (int c = threadIdx.x; c < 32; c += 64) {
                *(unsigned*)&Kt[rr][c * 2] = 0u;
                *(unsigned*)&Vt[rr][c * 2] = 0u;
            }
        }
    }
    long tok0 = (long)nb * NQ + lo - 48;
#if TDM_AVAILABLE
    if (mt == 0)
        tdm_load_2d((unsigned)(unsigned long long)&Kt[lo][0],
                    kp + (tok0 * NH + h) * DP, DP, (unsigned)(hi - lo), NH * DP,
                    /*interval 32 dw*/4, /*amount 4 dw*/3, true);
    else
        tdm_load_2d((unsigned)(unsigned long long)&Vt[lo][0],
                    vp + (tok0 * NH + h) * DP, DP, (unsigned)(hi - lo), NH * DP,
                    4, 3, true);
    __builtin_amdgcn_s_wait_tensorcnt(0);
#else
    for (int e = threadIdx.x; e < (hi - lo) * 32; e += 64) {
        int rr = lo + (e >> 5), c = e & 31;
        long tok = (long)nb * NQ + rr - 48;
        *(unsigned*)&Kt[rr][c * 2] = *(const unsigned*)(kp + (tok * NH + h) * DP + c * 2);
        *(unsigned*)&Vt[rr][c * 2] = *(const unsigned*)(vp + (tok * NH + h) * DP + c * 2);
    }
#endif
    __syncthreads();

    long tokq = (long)nb * NQ + mt * 16 + ln;
    const unsigned short* qrow = qp + (tokq * NH + h) * DP;
    bfrag qa0 = load_frag_A(qrow, 0, lane);
    bfrag qa1 = load_frag_A(qrow, 32, lane);

    v8f S[8];
    const float* bptr = bias + ((long)nb * NH + h) * NQ * NKY;
#pragma unroll
    for (int nt = 0; nt < 8; ++nt) {
        int keyl = nt * 16 + ln;
        const unsigned short* kcol = &Kt[keyl][0];
        v8f acc = {};
        acc = wmma_bf16(qa0, load_frag_B(kcol, 0, lane), acc);
        acc = wmma_bf16(qa1, load_frag_B(kcol, 32, lane), acc);
        bool valid = (keyl >= lo) && (keyl < hi);
#pragma unroll
        for (int j = 0; j < 8; ++j) {
            int qr = mt * 16 + j + half * 8;
            float v = acc[j] + bptr[(long)qr * NKY + keyl];
            acc[j] = valid ? v : -1e10f;
        }
        S[nt] = acc;
    }

    // row softmax (rows live across 16 lanes of one half)
    float mx[8], sm[8];
#pragma unroll
    for (int j = 0; j < 8; ++j) {
        float m = S[0][j];
#pragma unroll
        for (int nt = 1; nt < 8; ++nt) m = fmaxf(m, S[nt][j]);
        for (int off = 1; off < 16; off <<= 1) m = fmaxf(m, __shfl_xor(m, off));
        mx[j] = m;
    }
#pragma unroll
    for (int j = 0; j < 8; ++j) {
        float ssum = 0.f;
#pragma unroll
        for (int nt = 0; nt < 8; ++nt) { float e = __expf(S[nt][j] - mx[j]); S[nt][j] = e; ssum += e; }
        for (int off = 1; off < 16; off <<= 1) ssum += __shfl_xor(ssum, off);
        sm[j] = 1.f / ssum;
    }
#pragma unroll
    for (int nt = 0; nt < 8; ++nt) {
        int keyl = nt * 16 + ln;
#pragma unroll
        for (int j = 0; j < 8; ++j)
            P[mt][j + half * 8][keyl] = f2bf(S[nt][j] * sm[j]);
    }
    __syncthreads();

    // O = P @ V (only d < 48 matters)
#pragma unroll
    for (int dt = 0; dt < 3; ++dt) {
        v8f acc = {};
#pragma unroll
        for (int ks = 0; ks < 4; ++ks) {
            bfrag pa = load_frag_A(&P[mt][ln][0], ks * 32, lane);
            FragU bf;
#pragma unroll
            for (int v = 0; v < 8; ++v) {
                int k2 = ks * 32 + half * 16 + 2 * v;
                unsigned lo16 = Vt[k2][dt * 16 + ln];
                unsigned hi16 = Vt[k2 + 1][dt * 16 + ln];
                bf.u[v] = lo16 | (hi16 << 16);
            }
            acc = wmma_bf16(pa, bf.b, acc);
        }
#pragma unroll
        for (int j = 0; j < 8; ++j) {
            int qr = mt * 16 + j + half * 8;
            long tok = (long)nb * NQ + qr;
            o[tok * CA + h * DH + dt * 16 + ln] = acc[j];
        }
    }
}

// go = bf16(g * o)
__global__ void k_gmul(const float* __restrict__ g, const float* __restrict__ o,
                       unsigned short* __restrict__ go) {
    int i = blockIdx.x * blockDim.x + threadIdx.x;
    if (i < N_TOK * CA) go[i] = f2bf(g[i] * o[i]);
}

// out = sigmoid(s@w_last + b_last) * (go@wo + bo). Each wave: 16 rows x 6 N-tiles.
__global__ void k_final(const unsigned short* __restrict__ go,    // [N][CA] bf16
                        const unsigned short* __restrict__ woT,   // [CA][CA] bf16
                        const float* __restrict__ bo,
                        const unsigned short* __restrict__ s_bf,  // [N][CS] bf16
                        const unsigned short* __restrict__ wlT,   // [CA][CS] bf16
                        const float* __restrict__ b_last,
                        float* __restrict__ out) {
    int lane = threadIdx.x & 31, wave = threadIdx.x >> 5;
    int job = blockIdx.x * 8 + wave;
    int tm = job >> 3, tg = job & 7;
    int half = lane >> 4, ln = lane & 15;
    long row = (long)tm * 16 + ln;
    v8f acc1[6] = {}, acc2[6] = {};
    const unsigned short* arow1 = go + row * CA;
    for (int kk = 0; kk < CA; kk += 32) {
        bfrag af = load_frag_A(arow1, kk, lane);
#pragma unroll
        for (int t = 0; t < 6; ++t)
            acc1[t] = wmma_bf16(af, load_frag_B(woT + (long)((tg * 6 + t) * 16 + ln) * CA, kk, lane),
                                acc1[t]);
    }
    const unsigned short* arow2 = s_bf + row * CS;
    for (int kk = 0; kk < CS; kk += 32) {
        bfrag af = load_frag_A(arow2, kk, lane);
#pragma unroll
        for (int t = 0; t < 6; ++t)
            acc2[t] = wmma_bf16(af, load_frag_B(wlT + (long)((tg * 6 + t) * 16 + ln) * CS, kk, lane),
                                acc2[t]);
    }
#pragma unroll
    for (int t = 0; t < 6; ++t) {
        int col = (tg * 6 + t) * 16 + ln;
        float bob = bo[col], blb = b_last[col];
#pragma unroll
        for (int j = 0; j < 8; ++j) {
            long r = (long)tm * 16 + j + half * 8;
            out[r * CA + col] = sigm(acc2[t][j] + blb) * (acc1[t][j] + bob);
        }
    }
}

// ---------------------------------------------------------------------------
// Host launcher
// ---------------------------------------------------------------------------
extern "C" void kernel_launch(void* const* d_in, const int* in_sizes, int n_in,
                              void* d_out, int out_size, void* d_ws, size_t ws_size,
                              hipStream_t stream) {
    const float* a         = (const float*)d_in[0];
    const float* s         = (const float*)d_in[1];
    const float* z         = (const float*)d_in[2];
    const float* adaln_g_w = (const float*)d_in[3];
    const float* adaln_g_b = (const float*)d_in[4];
    const float* adaln_s_w = (const float*)d_in[5];
    const float* gz        = (const float*)d_in[6];
    const float* bz        = (const float*)d_in[7];
    const float* wz        = (const float*)d_in[8];
    const float* wq        = (const float*)d_in[9];
    const float* wk        = (const float*)d_in[10];
    const float* wv        = (const float*)d_in[11];
    const float* wg        = (const float*)d_in[12];
    const float* bg        = (const float*)d_in[13];
    const float* wo        = (const float*)d_in[14];
    const float* bo        = (const float*)d_in[15];
    const float* w_last    = (const float*)d_in[16];
    const float* b_last    = (const float*)d_in[17];
    float* out = (float*)d_out;

    char* ws = (char*)d_ws;
    size_t off = 0;
    auto alloc = [&](size_t sz) -> char* {
        char* p = ws + off; off = (off + sz + 255) & ~(size_t)255; return p;
    };
    unsigned short* sln_bf = (unsigned short*)alloc((size_t)N_TOK * CS * 2);
    unsigned short* s_bf   = (unsigned short*)alloc((size_t)N_TOK * CS * 2);
    float*          aln_f  = (float*)         alloc((size_t)N_TOK * CA * 4);
    unsigned short* a2_bf  = (unsigned short*)alloc((size_t)N_TOK * CA * 2);
    unsigned short* qp     = (unsigned short*)alloc((size_t)N_TOK * NH * DP * 2);
    unsigned short* kp     = (unsigned short*)alloc((size_t)N_TOK * NH * DP * 2);
    unsigned short* vp     = (unsigned short*)alloc((size_t)N_TOK * NH * DP * 2);
    float*          g_f    = (float*)         alloc((size_t)N_TOK * CA * 4);
    float*          bias_f = (float*)         alloc((size_t)NB * NH * NQ * NKY * 4);
    float*          o_f    = (float*)         alloc((size_t)N_TOK * CA * 4);
    unsigned short* go_bf  = (unsigned short*)alloc((size_t)N_TOK * CA * 2);
    unsigned short* wT_g   = (unsigned short*)alloc((size_t)CA * CS * 2);
    unsigned short* wT_s   = (unsigned short*)alloc((size_t)CA * CS * 2);
    unsigned short* wT_q   = (unsigned short*)alloc((size_t)CA * CA * 2);
    unsigned short* wT_k   = (unsigned short*)alloc((size_t)CA * CA * 2);
    unsigned short* wT_v   = (unsigned short*)alloc((size_t)CA * CA * 2);
    unsigned short* wT_gt  = (unsigned short*)alloc((size_t)CA * CA * 2);
    unsigned short* wT_o   = (unsigned short*)alloc((size_t)CA * CA * 2);
    unsigned short* wT_l   = (unsigned short*)alloc((size_t)CA * CS * 2);
    unsigned short* wzT    = (unsigned short*)alloc((size_t)NH * CZ * 2);

    struct TJob { const float* src; unsigned short* dst; int K; int N; };
    TJob jobs[9] = {
        { adaln_g_w, wT_g, CS, CA }, { adaln_s_w, wT_s, CS, CA },
        { wq, wT_q, CA, CA }, { wk, wT_k, CA, CA }, { wv, wT_v, CA, CA },
        { wg, wT_gt, CA, CA }, { wo, wT_o, CA, CA }, { w_last, wT_l, CS, CA },
        { wz, wzT, CZ, NH },
    };
    for (int i = 0; i < 9; ++i) {
        int n = jobs[i].K * jobs[i].N;
        k_transpose_bf16<<<(n + 255) / 256, 256, 0, stream>>>(jobs[i].src, jobs[i].dst,
                                                              jobs[i].K, jobs[i].N);
    }

    hipMemsetAsync(qp, 0, (size_t)N_TOK * NH * DP * 2, stream);
    hipMemsetAsync(kp, 0, (size_t)N_TOK * NH * DP * 2, stream);
    hipMemsetAsync(vp, 0, (size_t)N_TOK * NH * DP * 2, stream);

    k_ln_s<<<N_TOK, 128, 0, stream>>>(s, sln_bf, s_bf);
    k_ln_a<<<N_TOK, 256, 0, stream>>>(a, aln_f);

    k_adaln<<<512, 256, 0, stream>>>(sln_bf, wT_g, wT_s, adaln_g_b, aln_f, a2_bf);
    k_qkvg<<<dim3(256, 4), 256, 0, stream>>>(a2_bf, wT_q, wT_k, wT_v, wT_gt, bg,
                                             qp, kp, vp, g_f);
    k_bias<<<(NB * NQ * NKY) / 16, 256, 0, stream>>>(z, gz, bz, wzT, bias_f);
    k_attn<<<dim3(NB, NH), 64, 0, stream>>>(qp, kp, vp, bias_f, o_f);
    k_gmul<<<(N_TOK * CA + 255) / 256, 256, 0, stream>>>(g_f, o_f, go_bf);
    k_final<<<256, 256, 0, stream>>>(go_bf, wT_o, bo, s_bf, wT_l, b_last, out);
}
